// Sequence_Time_LSTM_1_59889023976101
// MI455X (gfx1250) — matrix operand order repaired and checked by reading, not compiled, not run on hardware
//
#include <hip/hip_runtime.h>

#define SEQ   256
#define BATCH 4096
#define HID   256
#define BT    16          // batch rows per workgroup
#define LDA   808         // sA row stride (bf16): [x 0..31 | hs1 32..287 | hA 288..543 | hB 544..799] + pad
#define LDT   264         // time-gate buffer stride (bank-staggered)
#define LDD   40          // demb buffer stride

typedef __attribute__((ext_vector_type(16))) __bf16 v16bf;
typedef __attribute__((ext_vector_type(8)))  __bf16 v8bf;
typedef __attribute__((ext_vector_type(8)))  float  v8f;

// ---- bf16 packed-weight workspace layout (element offsets) ----
#define WP_OFF   0u        // WpT  [1024][544]  K-order = [x(32,pad from 20) | W_s_p(256) | W_h_p(256)]
#define WS_OFF   557056u   // WsT  [1024][32]   W_x_s^T, K padded 20->32
#define WXT_OFF  589824u   // WxtT [256][32]    W_xt^T, K padded 20->32
#define WDT_OFF  598016u   // WdtT [256][32]    W_dt^T
#define WM_OFF   606208u   // WmT  [32][256]    Wm^T, N padded 22->32
#define WC_OFF   614400u   // WcT  [16][256]    Wc^T, N padded 3->16
#define WS_TOTAL 618496u   // ~1.24 MB bf16 total

// Fast transcendentals: single v_exp_f32 + v_rcp_f32 (TRANS ops co-execute with XDL WMMA),
// instead of the IEEE div_scale/div_fixup ladder libm division emits.
__device__ __forceinline__ float sigm(float x){
  return __builtin_amdgcn_rcpf(1.0f + __expf(-x));          // saturates correctly at +-inf
}
__device__ __forceinline__ float tanh_fast(float x){
  return 1.0f - 2.0f*__builtin_amdgcn_rcpf(1.0f + __expf(2.0f*x));  // sign-correct, -> +-1
}

// A fragment (16x32 bf16, row-major in LDS). ISA layout: lanes 0-15 hold K 0..7,16..23;
// lanes 16-31 hold K 8..15,24..31 for row M=lane&15.
__device__ __forceinline__ v16bf ldA(const __bf16* base, int lda, int kbase, int lane){
  int m = lane & 15, half = lane >> 4;
  const __bf16* p = base + m*lda + kbase + half*8;
  v8bf lo = *(const v8bf*)p;
  v8bf hi = *(const v8bf*)(p + 16);
  return __builtin_shufflevector(lo, hi, 0,1,2,3,4,5,6,7,8,9,10,11,12,13,14,15);
}
// B fragment (32x16 bf16). Weights stored transposed (N-major, ldk elems per col),
// col n = lane&15; lanes 0-15 hold K 0..7,16..23, lanes 16-31 hold K 8..15,24..31 (matches ldA).
__device__ __forceinline__ v16bf ldB(const __bf16* wt, int ldk, int ncol0, int kbase, int lane){
  int n = ncol0 + (lane & 15), ko = (lane >> 4)*8;
  const __bf16* p = wt + (size_t)n*ldk + kbase + ko;
  v8bf lo = *(const v8bf*)p;
  v8bf hi = *(const v8bf*)(p + 16);
  return __builtin_shufflevector(lo, hi, 0,1,2,3,4,5,6,7,8,9,10,11,12,13,14,15);
}
__device__ __forceinline__ v8f wmma_bf16(v16bf a, v16bf b, v8f c){
  return __builtin_amdgcn_wmma_f32_16x16x32_bf16(false, a, false, b, (short)0, c, false, false);
}

// ---------------- weight packing: fp32 -> transposed bf16 ----------------
__global__ void tlstm_prep(const float* __restrict__ W_xt, const float* __restrict__ W_dt,
                           const float* __restrict__ W_x_s, const float* __restrict__ W_x_p,
                           const float* __restrict__ W_h_p, const float* __restrict__ W_s_p,
                           const float* __restrict__ Wm,   const float* __restrict__ Wc,
                           __bf16* __restrict__ ws)
{
  unsigned i = blockIdx.x*256u + threadIdx.x;
  if (i >= WS_TOTAL) return;
  float v = 0.0f;
  if (i < WS_OFF){                                   // WpT [1024][544]
    unsigned n = i/544u, k = i%544u;
    if      (k <  32u) v = (k < 20u) ? W_x_p[k*1024u + n] : 0.0f;
    else if (k < 288u) v = W_s_p[(k-32u)*1024u + n];
    else               v = W_h_p[(k-288u)*1024u + n];
  } else if (i < WXT_OFF){                           // WsT [1024][32]
    unsigned j = i-WS_OFF, n = j/32u, k = j%32u;
    v = (k < 20u) ? W_x_s[k*1024u + n] : 0.0f;
  } else if (i < WDT_OFF){                           // WxtT [256][32]
    unsigned j = i-WXT_OFF, n = j/32u, k = j%32u;
    v = (k < 20u) ? W_xt[k*256u + n] : 0.0f;
  } else if (i < WM_OFF){                            // WdtT [256][32]
    unsigned j = i-WDT_OFF, n = j/32u, k = j%32u;
    v = W_dt[k*256u + n];
  } else if (i < WC_OFF){                            // WmT [32][256]
    unsigned j = i-WM_OFF, n = j/256u, k = j%256u;
    v = (n < 22u) ? Wm[k*22u + n] : 0.0f;
  } else {                                           // WcT [16][256]
    unsigned j = i-WC_OFF, n = j/256u, k = j%256u;
    v = (n < 3u) ? Wc[k*3u + n] : 0.0f;
  }
  ws[i] = (__bf16)v;
}

// ---------------- persistent Time-LSTM scan kernel ----------------
__global__ __launch_bounds__(256, 1) void tlstm_main(
    const float* __restrict__ val_seq, const float* __restrict__ delta,
    const float* __restrict__ W_d,  const float* __restrict__ b_d,
    const float* __restrict__ b_t,  const float* __restrict__ b_s,
    const float* __restrict__ b_p,  const float* __restrict__ bc,
    const float* __restrict__ bm,   const __bf16* __restrict__ ws,
    float* __restrict__ out)
{
  __shared__ __bf16 sA[BT][LDA];     // activations: x | hs1 | h (ping) | h (pong)
  __shared__ __bf16 sT[BT][LDT];     // time gate T
  __shared__ __bf16 sD[BT][LDD];     // delta embedding (E=32)
  __shared__ float  sLog[BT][48];    // raw heads: val logits 0..31, cat logits 32..47

  const int tid  = threadIdx.x;
  const int lane = tid & 31, wv = tid >> 5;
  const int rowg0 = blockIdx.x * BT;

  const __bf16* WpT  = ws + WP_OFF;
  const __bf16* WsT  = ws + WS_OFF;
  const __bf16* WxtT = ws + WXT_OFF;
  const __bf16* WdtT = ws + WDT_OFF;
  const __bf16* WmT  = ws + WM_OFF;
  const __bf16* WcT  = ws + WC_OFF;

  float* out_cat = out;
  float* out_val = out + (size_t)(SEQ-1)*BATCH*3;

  // init: x <- val_seq[0] (cols 0..19); pads, hs1 and both h buffers zeroed (h0 = 0)
  for (int i = tid; i < BT*LDA; i += 256){
    int r = i / LDA, c = i % LDA;
    float v = 0.0f;
    if (c < 20) v = val_seq[(size_t)(rowg0 + r)*20 + c];
    sA[r][c] = (__bf16)v;
  }
  v8f cst[2];                                  // private cell state c, per-wave resident
  for (int q = 0; q < 2; ++q) for (int r = 0; r < 8; ++r) cst[q][r] = 0.0f;
  __syncthreads();

  for (int t = 0; t < SEQ-1; ++t){
    const int hRead  = 288 + ((t    ) & 1)*256;   // ping-pong h inside sA
    const int hWrite = 288 + ((t + 1) & 1)*256;

    // ---- A: demb = tanh(d * W_d + b_d) ----
    {
      int r = tid >> 4, e0 = (tid & 15)*2;
      float d = delta[(size_t)t*BATCH + rowg0 + r];
      sD[r][e0]   = (__bf16)tanh_fast(d*W_d[e0]   + b_d[e0]);
      sD[r][e0+1] = (__bf16)tanh_fast(d*W_d[e0+1] + b_d[e0+1]);
    }
    __syncthreads();

    // ---- T: T = sigmoid(x@W_xt + sigmoid(demb@W_dt) + b_t); sigmoid(u) fused via C operand ----
    for (int q = 0; q < 2; ++q){
      int nt = wv*2 + q;
      v8f z = {0,0,0,0,0,0,0,0};
      v8f u = wmma_bf16(ldA(&sD[0][0], LDD, 0, lane), ldB(WdtT, 32, nt*16, 0, lane), z);
      for (int r = 0; r < 8; ++r) u[r] = sigm(u[r]);
      v8f tp = wmma_bf16(ldA(&sA[0][0], LDA, 0, lane), ldB(WxtT, 32, nt*16, 0, lane), u);
      int nH = nt*16 + (lane & 15), rb = (lane >> 4)*8;
      float btv = b_t[nH];
      for (int r = 0; r < 8; ++r) sT[rb + r][nH] = (__bf16)sigm(tp[r] + btv);
    }
    __syncthreads();

    // ---- S: shared path (hs=cs=0): hs1 = sig(o)*tanh(sig(i)*T*tanh(g)) ----
    for (int q = 0; q < 2; ++q){
      int nt = wv*2 + q;
      v16bf aX = ldA(&sA[0][0], LDA, 0, lane);
      v8f g4[4];
      #pragma unroll
      for (int g = 0; g < 4; ++g){
        v8f z = {0,0,0,0,0,0,0,0};
        g4[g] = wmma_bf16(aX, ldB(WsT, 32, g*HID + nt*16, 0, lane), z);
      }
      int nH = nt*16 + (lane & 15), rb = (lane >> 4)*8;
      float bi = b_s[nH], bo = b_s[2*HID + nH], bg = b_s[3*HID + nH];
      for (int r = 0; r < 8; ++r){
        float Tv = (float)sT[rb + r][nH];
        float cn = sigm(g4[0][r] + bi) * Tv * tanh_fast(g4[3][r] + bg);
        sA[rb + r][32 + nH] = (__bf16)(sigm(g4[2][r] + bo) * tanh_fast(cn));
      }
    }
    __syncthreads();

    // ---- P: private path: pre = [x | hs1 | h] @ WpT + b_p  (K = 544 = 17 WMMA steps) ----
    for (int q = 0; q < 2; ++q){
      int nt = wv*2 + q;
      v8f acc[4];
      #pragma unroll
      for (int g = 0; g < 4; ++g) for (int r = 0; r < 8; ++r) acc[g][r] = 0.0f;
      for (int kt = 0; kt < 17; ++kt){
        int acol = (kt < 9) ? kt*32 : hRead + (kt-9)*32;   // x|hs1 contiguous, h ping-pong
        v16bf a = ldA(&sA[0][0], LDA, acol, lane);
        #pragma unroll
        for (int g = 0; g < 4; ++g)
          acc[g] = wmma_bf16(a, ldB(WpT, 544, g*HID + nt*16, kt*32, lane), acc[g]);
      }
      int nH = nt*16 + (lane & 15), rb = (lane >> 4)*8;
      float bi = b_p[nH], bf = b_p[HID + nH], bo = b_p[2*HID + nH], bg = b_p[3*HID + nH];
      for (int r = 0; r < 8; ++r){
        float Tv = (float)sT[rb + r][nH];
        float cn = sigm(acc[1][r] + bf) * cst[q][r]
                 + sigm(acc[0][r] + bi) * Tv * tanh_fast(acc[3][r] + bg);
        cst[q][r] = cn;
        sA[rb + r][hWrite + nH] = (__bf16)(sigm(acc[2][r] + bo) * tanh_fast(cn));
      }
    }
    __syncthreads();

    // ---- O: heads o_val = h1@Wm+bm (N->32), cat logits = h1@Wc+bc (N->16) ----
    if (wv < 3){
      bool isCat = (wv == 2);
      int nt = isCat ? 0 : wv;
      const __bf16* W = isCat ? WcT : WmT;
      v8f acc = {0,0,0,0,0,0,0,0};
      for (int kt = 0; kt < 8; ++kt)
        acc = wmma_bf16(ldA(&sA[0][0], LDA, hWrite + kt*32, lane),
                        ldB(W, 256, nt*16, kt*32, lane), acc);
      int n = nt*16 + (lane & 15), rb = (lane >> 4)*8;
      float bias = isCat ? ((n < 3) ? bc[n] : 0.0f) : ((n < 22) ? bm[n] : 0.0f);
      int cb = isCat ? 32 : 0;
      for (int r = 0; r < 8; ++r) sLog[rb + r][cb + n] = acc[r] + bias;
    }
    __syncthreads();

    // ---- C: emit outputs, softmax(cat), next x with NaN fill from o_val[:,2:] ----
    {
      int r = tid >> 4, j = tid & 15;
      size_t rg  = (size_t)rowg0 + r;
      size_t ovb = ((size_t)t*BATCH + rg)*22;
      #pragma unroll
      for (int s = 0; s < 2; ++s){
        int c = j + s*16;
        if (c < 22) out_val[ovb + c] = sLog[r][c];
        if (c < 20){
          float vn = val_seq[((size_t)(t+1)*BATCH + rg)*20 + c];
          if (vn != vn) vn = sLog[r][c + 2];        // isnan -> o_val[:,2:]
          sA[r][c] = (__bf16)vn;
        }
      }
      if (j == 0){
        float l0 = sLog[r][32], l1 = sLog[r][33], l2 = sLog[r][34];
        float mx = fmaxf(l0, fmaxf(l1, l2));
        float e0 = __expf(l0-mx), e1 = __expf(l1-mx), e2 = __expf(l2-mx);
        float inv = __builtin_amdgcn_rcpf(e0 + e1 + e2);
        size_t ocb = ((size_t)t*BATCH + rg)*3;
        out_cat[ocb]   = e0*inv;
        out_cat[ocb+1] = e1*inv;
        out_cat[ocb+2] = e2*inv;
      }
    }
    __syncthreads();
  }
}

extern "C" void kernel_launch(void* const* d_in, const int* in_sizes, int n_in,
                              void* d_out, int out_size, void* d_ws, size_t ws_size,
                              hipStream_t stream) {
  (void)in_sizes; (void)n_in; (void)out_size; (void)ws_size;
  // setup_inputs order: 0 cat_seq(dead) 1 val_seq 2 delta 3 W_d 4 b_d 5 W_xt 6 W_dt 7 b_t
  // 8 W_x_s 9 W_h_s(dead: hs=0) 10 b_s 11 W_x_p 12 W_h_p 13 W_s_p 14 b_p 15 Wc 16 bc 17 Wm 18 bm 19 task_name
  const float* val_seq = (const float*)d_in[1];
  const float* delta   = (const float*)d_in[2];
  const float* W_d     = (const float*)d_in[3];
  const float* b_d     = (const float*)d_in[4];
  const float* W_xt    = (const float*)d_in[5];
  const float* W_dt    = (const float*)d_in[6];
  const float* b_t     = (const float*)d_in[7];
  const float* W_x_s   = (const float*)d_in[8];
  const float* b_s     = (const float*)d_in[10];
  const float* W_x_p   = (const float*)d_in[11];
  const float* W_h_p   = (const float*)d_in[12];
  const float* W_s_p   = (const float*)d_in[13];
  const float* b_p     = (const float*)d_in[14];
  const float* Wc      = (const float*)d_in[15];
  const float* bc      = (const float*)d_in[16];
  const float* Wm      = (const float*)d_in[17];
  const float* bm      = (const float*)d_in[18];

  __bf16* ws = (__bf16*)d_ws;          // needs ~1.24 MB
  float*  out = (float*)d_out;

  tlstm_prep<<<(WS_TOTAL + 255u)/256u, 256, 0, stream>>>(
      W_xt, W_dt, W_x_s, W_x_p, W_h_p, W_s_p, Wm, Wc, ws);
  tlstm_main<<<BATCH/BT, 256, 0, stream>>>(
      val_seq, delta, W_d, b_d, b_t, b_s, b_p, bc, bm, ws, out);
}
